// VisionMamba_84799834292377
// MI455X (gfx1250) — compile-verified, matrix operand order checked by reading, not run on hardware
//
#include <hip/hip_runtime.h>
#include <math.h>

// ---------------- CDNA5 WMMA types ----------------
typedef __attribute__((ext_vector_type(16))) _Float16 v16h;
typedef __attribute__((ext_vector_type(8)))  _Float16 v8h;
typedef __attribute__((ext_vector_type(8)))  float    v8f;

#define MROWS  3136      // B * L  (= 49 * 64)
#define DM     384
#define REG    12        // 384 / 32 lanes
#define LTOK   196
#define BSZ    16
#define DIN    768
#define DST    16
#define DTR    24
#define DBLP   64        // padded row stride of dbl (was 56)
#define NDEPTH 24
#define NCLS   1000

#define SQC   0.70710678118654752f          // sqrt(0.5)
#define MAXN  (0.95f / SQC)

// ---------------- wave32 helpers ----------------
__device__ __forceinline__ float wsum(float v) {
#pragma unroll
  for (int m = 16; m > 0; m >>= 1) v += __shfl_xor(v, m, 32);
  return v;
}
__device__ __forceinline__ float atanh_(float x) {
  return 0.5f * logf((1.0f + x) / (1.0f - x));
}
__device__ __forceinline__ float silu_(float x) {
  return x / (1.0f + expf(-x));
}

// row = 384 floats, interleaved: lane holds dims d = e*32 + lane
__device__ __forceinline__ float row_norm(const float v[REG]) {
  float s = 0.f;
#pragma unroll
  for (int e = 0; e < REG; e++) s += v[e] * v[e];
  s = wsum(s);
  return sqrtf(s + 1e-15f);
}
__device__ __forceinline__ void row_expmap0(float v[REG]) {
  float n = fminf(fmaxf(row_norm(v), 1e-8f), 5.0f);
  float f = tanhf(SQC * n) / (SQC * n);
#pragma unroll
  for (int e = 0; e < REG; e++) v[e] *= f;
  float rn = row_norm(v);
  float sc = rn > MAXN ? MAXN / rn : 1.0f;
#pragma unroll
  for (int e = 0; e < REG; e++) v[e] *= sc;
}
__device__ __forceinline__ void row_logmap0(float v[REG]) {
  float n = fminf(fmaxf(row_norm(v), 1e-8f), MAXN);
  float f = atanh_(fminf(SQC * n, 0.95f)) / (SQC * n);
#pragma unroll
  for (int e = 0; e < REG; e++) v[e] *= f;
}
__device__ __forceinline__ void row_hyp_ln(float v[REG], const float* g,
                                           const float* b, int lane) {
  row_logmap0(v);
  float s = 0.f;
#pragma unroll
  for (int e = 0; e < REG; e++) s += v[e];
  float mean = wsum(s) * (1.0f / DM);
#pragma unroll
  for (int e = 0; e < REG; e++) v[e] -= mean;
  float q = 0.f;
#pragma unroll
  for (int e = 0; e < REG; e++) q += v[e] * v[e];
  float var = wsum(q) * (1.0f / (DM - 1));
  float inv = 1.0f / (sqrtf(var) + 1e-5f);
#pragma unroll
  for (int e = 0; e < REG; e++) {
    int d = e * 32 + lane;
    v[e] = v[e] * inv * g[d] + b[d];
  }
  row_expmap0(v);
}

// ---------------- WMMA GEMM: C = act(A @ W^T + bias), f16 operands -------
// Invariants (guaranteed by caller): M%64==0, N%64==0, K%32==0,
// element strides %4==0, 16B-aligned pointers. One wave -> 64x16 output
// (4 M-tiles reuse one B fragment); block(32,4) -> 64x64 supertile.
#define ACT_NONE 0
#define ACT_SOFTPLUS 1

// Per-lane fragment: two contiguous 8-half runs at +0 and +16 (ISA 16-bit
// A/B layout, khalf folded into the pointer). 2x global_load_b128, no cvt.
__device__ __forceinline__ v16h load_frag(const _Float16* __restrict__ p) {
  v8h r0 = *(const v8h*)(p);
  v8h r1 = *(const v8h*)(p + 16);
  v16h f;
#pragma unroll
  for (int j = 0; j < 8; j++) {
    f[j]     = r0[j];
    f[j + 8] = r1[j];
  }
  return f;
}

__device__ __forceinline__ void store_tile(float* __restrict__ C,
                                           _Float16* __restrict__ C16,
                                           const v8f& acc, int mbase, int hi,
                                           int nc, int ldc, float bval, int act) {
#pragma unroll
  for (int r = 0; r < 8; r++) {
    float v = acc[r] + bval;
    if (act == ACT_SOFTPLUS) v = (v > 20.f) ? v : log1pf(expf(v));
    size_t i = (size_t)(mbase + r + hi * 8) * ldc + nc;
    C[i] = v;
    if (C16) C16[i] = (_Float16)v;
  }
}

__global__ __launch_bounds__(128)
void k_gemm_wmma(const _Float16* __restrict__ A, const _Float16* __restrict__ W,
                 const float* __restrict__ bias, float* __restrict__ C,
                 _Float16* __restrict__ C16,
                 int K, int lda, int ldw, int ldc, int act) {
  const int lane  = threadIdx.x;               // 0..31
  const int n0    = (blockIdx.x * 4 + threadIdx.y) * 16;
  const int m0    = blockIdx.y * 64;
  const int hi    = lane >> 4;                 // 0/1
  const int khalf = hi << 3;
  const int mr    = lane & 15;

  const _Float16* aptr = A + (size_t)(m0 + mr) * lda + khalf;
  const _Float16* wptr = W + (size_t)(n0 + mr) * ldw + khalf;
  __builtin_prefetch(wptr, 0, 1);              // global_prefetch_b8

  v8f acc0 = {}, acc1 = {}, acc2 = {}, acc3 = {};
  for (int kk = 0; kk < K; kk += 32) {
    v16h b  = load_frag(wptr + kk);
    v16h a0 = load_frag(aptr + kk);
    v16h a1 = load_frag(aptr + (size_t)16 * lda + kk);
    v16h a2 = load_frag(aptr + (size_t)32 * lda + kk);
    v16h a3 = load_frag(aptr + (size_t)48 * lda + kk);
    acc0 = __builtin_amdgcn_wmma_f32_16x16x32_f16(false, a0, false, b, (short)0, acc0, false, false);
    acc1 = __builtin_amdgcn_wmma_f32_16x16x32_f16(false, a1, false, b, (short)0, acc1, false, false);
    acc2 = __builtin_amdgcn_wmma_f32_16x16x32_f16(false, a2, false, b, (short)0, acc2, false, false);
    acc3 = __builtin_amdgcn_wmma_f32_16x16x32_f16(false, a3, false, b, (short)0, acc3, false, false);
  }

  const int nc = n0 + mr;
  const float bval = bias ? bias[nc] : 0.f;
  store_tile(C, C16, acc0, m0,      hi, nc, ldc, bval, act);
  store_tile(C, C16, acc1, m0 + 16, hi, nc, ldc, bval, act);
  store_tile(C, C16, acc2, m0 + 32, hi, nc, ldc, bval, act);
  store_tile(C, C16, acc3, m0 + 48, hi, nc, ldc, bval, act);
}

// ---------------- f32 -> f16 weight conversion ----------------
__global__ void k_f32_to_f16(const float* __restrict__ src, _Float16* __restrict__ dst, int n) {
  int idx = blockIdx.x * blockDim.x + threadIdx.x;
  if (idx < n) dst[idx] = (_Float16)src[idx];
}

// ---------------- weight padding prep (f16 out) ----------------
// xp_pad: [24][64][768], rows 56..63 zero   (so dbl pad cols compute to 0)
// dt_pad: [24][768][32], cols 24..31 zero   (so K-pad contributes 0)
__global__ void k_pad_weights(const float* __restrict__ xp_w, const float* __restrict__ dt_w,
                              _Float16* __restrict__ xp_pad, _Float16* __restrict__ dt_pad) {
  int idx = blockIdx.x * blockDim.x + threadIdx.x;
  const int xp_total = NDEPTH * 64 * DIN;
  const int dt_total = NDEPTH * DIN * 32;
  if (idx < xp_total) {
    int i = idx / (64 * DIN), r = (idx / DIN) % 64, k = idx % DIN;
    xp_pad[idx] = (r < 56) ? (_Float16)xp_w[((size_t)i * 56 + r) * DIN + k] : (_Float16)0.f;
  } else if (idx < xp_total + dt_total) {
    int j = idx - xp_total;
    int i = j / (DIN * 32), r = (j / 32) % DIN, k = j % 32;
    dt_pad[j] = (k < DTR) ? (_Float16)dt_w[((size_t)i * DIN + r) * DTR + k] : (_Float16)0.f;
  }
}

// ---------------- im2col for patch embedding (f16 out, GEMM-only) ---------
__global__ void k_im2col(const float* __restrict__ x, _Float16* __restrict__ out) {
  int idx = blockIdx.x * blockDim.x + threadIdx.x;
  if (idx >= MROWS * 768) return;
  int mrow = idx / 768, k = idx % 768;
  int b = mrow / LTOK, l = mrow % LTOK;
  int py = l / 14, px = l % 14;
  int ci = k >> 8, r = k & 255, iy = r >> 4, ix = r & 15;
  out[idx] = (_Float16)x[(((size_t)b * 3 + ci) * 224 + (py * 16 + iy)) * 224 + (px * 16 + ix)];
}

// ---------------- expmap0 over rows (in place f32 + f16 copy) ----------------
__global__ void k_expmap0_rows(float* __restrict__ h, _Float16* __restrict__ h16) {
  int row = blockIdx.x * 4 + threadIdx.y;
  if (row >= MROWS) return;
  int lane = threadIdx.x;
  float v[REG];
#pragma unroll
  for (int e = 0; e < REG; e++) v[e] = h[(size_t)row * DM + e * 32 + lane];
  row_expmap0(v);
#pragma unroll
  for (int e = 0; e < REG; e++) {
    size_t i = (size_t)row * DM + e * 32 + lane;
    h[i] = v[e];
    h16[i] = (_Float16)v[e];
  }
}

// ---------------- hyperbolic embedding epilogue ----------------
__global__ void k_embed_post(const float* __restrict__ hA, const float* __restrict__ mx,
                             const float* __restrict__ hyp_b,
                             const float* __restrict__ pe_g, const float* __restrict__ pe_b,
                             const float* __restrict__ pos, float* __restrict__ hidden) {
  int row = blockIdx.x * 4 + threadIdx.y;
  if (row >= MROWS) return;
  int lane = threadIdx.x;
  float a[REG], m[REG];
#pragma unroll
  for (int e = 0; e < REG; e++) {
    a[e] = hA[(size_t)row * DM + e * 32 + lane];
    m[e] = mx[(size_t)row * DM + e * 32 + lane];
  }
  float xn  = fmaxf(row_norm(a), 1e-8f);
  float mxn = fmaxf(row_norm(m), 1e-8f);
  float t   = tanhf(mxn / xn * atanh_(fminf(SQC * xn, 1.0f - 1e-5f)));
  float f   = t / (mxn * SQC);
#pragma unroll
  for (int e = 0; e < REG; e++) m[e] *= f;

  float bb[REG];
#pragma unroll
  for (int e = 0; e < REG; e++) bb[e] = hyp_b[e * 32 + lane];
  float xy = 0.f, x2 = 0.f, y2 = 0.f;
#pragma unroll
  for (int e = 0; e < REG; e++) { xy += m[e] * bb[e]; x2 += m[e] * m[e]; y2 += bb[e] * bb[e]; }
  xy = wsum(xy); x2 = wsum(x2); y2 = wsum(y2);
  float al  = 1.0f + xy + 0.5f * y2;         // 1 + 2c*xy + c*y2, c=0.5
  float be  = 1.0f - 0.5f * x2;
  float den = fmaxf(1.0f + xy + 0.25f * x2 * y2, 1e-15f);
#pragma unroll
  for (int e = 0; e < REG; e++) m[e] = (al * m[e] + be * bb[e]) / den;

  row_hyp_ln(m, pe_g, pe_b, lane);
  row_logmap0(m);
  int l = row % LTOK;
#pragma unroll
  for (int e = 0; e < REG; e++) m[e] += pos[(size_t)l * DM + e * 32 + lane];
  row_expmap0(m);
#pragma unroll
  for (int e = 0; e < REG; e++) hidden[(size_t)row * DM + e * 32 + lane] = m[e];
}

// ---------------- residual + hyp_layernorm (f32 + f16 outputs) -------------
__global__ void k_resid_ln(const float* __restrict__ hidden, float* __restrict__ resid,
                           const float* __restrict__ g, const float* __restrict__ b,
                           float* __restrict__ hn, _Float16* __restrict__ hn16, int first) {
  int row = blockIdx.x * 4 + threadIdx.y;
  if (row >= MROWS) return;
  int lane = threadIdx.x;
  float v[REG];
#pragma unroll
  for (int e = 0; e < REG; e++) {
    size_t i = (size_t)row * DM + e * 32 + lane;
    float h = hidden[i];
    float r = first ? h : (h + resid[i]);
    resid[i] = r;
    v[e] = r;
  }
  row_hyp_ln(v, g, b, lane);
#pragma unroll
  for (int e = 0; e < REG; e++) {
    size_t i = (size_t)row * DM + e * 32 + lane;
    hn[i] = v[e];
    hn16[i] = (_Float16)v[e];
  }
}

// ---------------- logmap0 over rows (in place) ----------------
__global__ void k_logmap0_rows(float* __restrict__ h) {
  int row = blockIdx.x * 4 + threadIdx.y;
  if (row >= MROWS) return;
  int lane = threadIdx.x;
  float v[REG];
#pragma unroll
  for (int e = 0; e < REG; e++) v[e] = h[(size_t)row * DM + e * 32 + lane];
  row_logmap0(v);
#pragma unroll
  for (int e = 0; e < REG; e++) h[(size_t)row * DM + e * 32 + lane] = v[e];
}

// ---------------- depthwise causal conv (width 4) + SiLU -------------------
__global__ void k_conv_silu(const float* __restrict__ xz, const float* __restrict__ cw,
                            const float* __restrict__ cb, float* __restrict__ u,
                            _Float16* __restrict__ u16) {
  int idx = blockIdx.x * blockDim.x + threadIdx.x;
  if (idx >= MROWS * DIN) return;
  int mrow = idx / DIN, c = idx % DIN;
  int b = mrow / LTOK, l = mrow % LTOK;
  float s = cb[c];
#pragma unroll
  for (int k = 0; k < 4; k++) {
    int ll = l + k - 3;
    if (ll >= 0) s += cw[c * 4 + k] * xz[(size_t)(b * LTOK + ll) * (2 * DIN) + c];
  }
  float r = silu_(s);
  u[idx] = r;
  u16[idx] = (_Float16)r;
}

// ---------------- selective scan (sequential over L, per (b,c) chain) ------
__global__ void k_scan(const float* __restrict__ u, const float* __restrict__ dt,
                       const float* __restrict__ dbl, const float* __restrict__ xz,
                       const float* __restrict__ A_log, const float* __restrict__ Dp,
                       _Float16* __restrict__ y16) {
  int idx = blockIdx.x * blockDim.x + threadIdx.x;
  if (idx >= BSZ * DIN) return;
  int b = idx / DIN, c = idx % DIN;
  float As[DST];
#pragma unroll
  for (int s = 0; s < DST; s++) As[s] = -expf(A_log[(size_t)c * DST + s]);
  float h[DST];
#pragma unroll
  for (int s = 0; s < DST; s++) h[s] = 0.f;
  const float Dc = Dp[c];
  for (int l = 0; l < LTOK; l++) {
    const size_t row = (size_t)b * LTOK + l;
    const float dtv = dt[row * DIN + c];
    const float uv  = u[row * DIN + c];
    const float* dr = dbl + row * DBLP;
    float acc = 0.f;
#pragma unroll
    for (int s = 0; s < DST; s++) {
      float dA = expf(dtv * As[s]);
      h[s] = dA * h[s] + dtv * dr[DTR + s] * uv;       // B part
      acc += h[s] * dr[DTR + DST + s];                 // C part
    }
    float z = xz[row * (2 * DIN) + DIN + c];
    y16[row * DIN + c] = (_Float16)((acc + Dc * uv) * silu_(z));
  }
}

// ---------------- pooled = expmap0(mean_l logmap0-rows) ----------------
__global__ void k_pool(const float* __restrict__ tlog, float* __restrict__ pooled) {
  int b = blockIdx.x;
  int lane = threadIdx.x;
  float v[REG];
#pragma unroll
  for (int e = 0; e < REG; e++) {
    float s = 0.f;
    int d = e * 32 + lane;
    for (int l = 0; l < LTOK; l++) s += tlog[(size_t)(b * LTOK + l) * DM + d];
    v[e] = s * (1.0f / LTOK);
  }
  row_expmap0(v);
#pragma unroll
  for (int e = 0; e < REG; e++) pooled[(size_t)b * DM + e * 32 + lane] = v[e];
}

// ---------------- hyperbolic distance head (closed-form norm) --------------
__global__ void k_classifier(const float* __restrict__ pooled,
                             const float* __restrict__ protos, float* __restrict__ out) {
  int idx = blockIdx.x * blockDim.x + threadIdx.x;
  if (idx >= BSZ * NCLS) return;
  int b = idx / NCLS, p = idx % NCLS;
  const float* a = pooled + (size_t)b * DM;
  const float* q = protos + (size_t)p * DM;
  float xy = 0.f, x2 = 0.f, y2 = 0.f;
  for (int d = 0; d < DM; d++) {
    float av = -a[d], qv = q[d];
    xy += av * qv; x2 += av * av; y2 += qv * qv;
  }
  float al  = 1.0f + xy + 0.5f * y2;
  float be  = 1.0f - 0.5f * x2;
  float den = fmaxf(1.0f + xy + 0.25f * x2 * y2, 1e-15f);
  float nn  = al * al * x2 + 2.0f * al * be * xy + be * be * y2;
  float dn  = sqrtf(fmaxf(nn, 0.f) + 1e-15f) / den;
  dn = fminf(SQC * dn, 1.0f - 1e-5f);
  out[(size_t)b * NCLS + p] = -(2.0f / SQC) * atanh_(dn);
}

// ---------------- host orchestration ----------------
static inline void gemm(hipStream_t st, const _Float16* A, const _Float16* W,
                        const float* bias, float* C, _Float16* C16,
                        int M, int N, int K, int lda, int ldw, int ldc, int act) {
  dim3 blk(32, 4);
  dim3 grd(N / 64, M / 64);
  hipLaunchKernelGGL(k_gemm_wmma, grd, blk, 0, st, A, W, bias, C, C16, K, lda, ldw, ldc, act);
}
static inline void cvt(hipStream_t st, const float* src, _Float16* dst, size_t n) {
  hipLaunchKernelGGL(k_f32_to_f16, dim3((n + 255) / 256), dim3(256), 0, st, src, dst, (int)n);
}

extern "C" void kernel_launch(void* const* d_in, const int* in_sizes, int n_in,
                              void* d_out, int out_size, void* d_ws, size_t ws_size,
                              hipStream_t stream) {
  const float* x         = (const float*)d_in[0];
  const float* patch_w   = (const float*)d_in[1];
  const float* patch_b   = (const float*)d_in[2];
  const float* hyp_w     = (const float*)d_in[3];
  const float* hyp_b     = (const float*)d_in[4];
  const float* pe_gamma  = (const float*)d_in[5];
  const float* pe_beta   = (const float*)d_in[6];
  const float* pos_embed = (const float*)d_in[7];
  const float* in_proj_w = (const float*)d_in[8];
  const float* conv_w    = (const float*)d_in[9];
  const float* conv_b    = (const float*)d_in[10];
  const float* x_proj_w  = (const float*)d_in[11];
  const float* dt_proj_w = (const float*)d_in[12];
  const float* dt_proj_b = (const float*)d_in[13];
  const float* A_log     = (const float*)d_in[14];
  const float* Dp        = (const float*)d_in[15];
  const float* out_proj_w= (const float*)d_in[16];
  const float* gamma     = (const float*)d_in[17];
  const float* beta      = (const float*)d_in[18];
  const float* gamma_f   = (const float*)d_in[19];
  const float* beta_f    = (const float*)d_in[20];
  const float* protos    = (const float*)d_in[21];
  float* out = (float*)d_out;

  // 256B-aligned workspace carving
  char* base = (char*)d_ws;
  size_t off = 0;
  auto alloc_f32 = [&](size_t n) { float* p = (float*)(base + off);
                                   off += ((n * 4 + 255) / 256) * 256; return p; };
  auto alloc_f16 = [&](size_t n) { _Float16* p = (_Float16*)(base + off);
                                   off += ((n * 2 + 255) / 256) * 256; return p; };

  float* xz      = alloc_f32((size_t)MROWS * 2 * DIN);
  float* u       = alloc_f32((size_t)MROWS * DIN);
  float* dtb     = alloc_f32((size_t)MROWS * DIN);
  float* dbl     = alloc_f32((size_t)MROWS * DBLP);
  float* hidden  = alloc_f32((size_t)MROWS * DM);
  float* resid   = alloc_f32((size_t)MROWS * DM);
  float* hn      = alloc_f32((size_t)MROWS * DM);
  float* mx      = alloc_f32((size_t)MROWS * DM);
  float* hA      = alloc_f32((size_t)MROWS * DM);
  float* pooled  = alloc_f32((size_t)BSZ * DM);

  _Float16* im16     = alloc_f16((size_t)MROWS * 768);
  _Float16* hA16     = alloc_f16((size_t)MROWS * DM);
  _Float16* hn16     = alloc_f16((size_t)MROWS * DM);
  _Float16* u16      = alloc_f16((size_t)MROWS * DIN);
  _Float16* dbl16    = alloc_f16((size_t)MROWS * DBLP);
  _Float16* yb16     = alloc_f16((size_t)MROWS * DIN);
  _Float16* wpatch16 = alloc_f16((size_t)DM * 768);
  _Float16* whyp16   = alloc_f16((size_t)DM * DM);
  _Float16* win16    = alloc_f16((size_t)NDEPTH * 2 * DIN * DM);
  _Float16* wout16   = alloc_f16((size_t)NDEPTH * DM * DIN);
  _Float16* xp_pad16 = alloc_f16((size_t)NDEPTH * 64 * DIN);
  _Float16* dt_pad16 = alloc_f16((size_t)NDEPTH * DIN * 32);
  (void)ws_size; (void)in_sizes; (void)n_in; (void)out_size;

  dim3 rblk(32, 4);
  dim3 rgrd((MROWS + 3) / 4);

  // 0. weight conversion / padding (bandwidth-trivial, once per call)
  cvt(stream, patch_w, wpatch16, (size_t)DM * 768);
  cvt(stream, hyp_w, whyp16, (size_t)DM * DM);
  cvt(stream, in_proj_w, win16, (size_t)NDEPTH * 2 * DIN * DM);
  cvt(stream, out_proj_w, wout16, (size_t)NDEPTH * DM * DIN);
  {
    int total = NDEPTH * 64 * DIN + NDEPTH * DIN * 32;
    hipLaunchKernelGGL(k_pad_weights, dim3((total + 255) / 256), dim3(256), 0, stream,
                       x_proj_w, dt_proj_w, xp_pad16, dt_pad16);
  }

  // 1. patch embed: im2col(f16) -> GEMM (+bias)
  hipLaunchKernelGGL(k_im2col, dim3((MROWS * 768 + 255) / 256), dim3(256), 0, stream, x, im16);
  gemm(stream, im16, wpatch16, patch_b, hA, nullptr, MROWS, DM, 768, 768, 768, DM, ACT_NONE);

  // 2. hyperbolic embedding
  hipLaunchKernelGGL(k_expmap0_rows, rgrd, rblk, 0, stream, hA, hA16);
  gemm(stream, hA16, whyp16, nullptr, mx, nullptr, MROWS, DM, DM, DM, DM, DM, ACT_NONE);
  hipLaunchKernelGGL(k_embed_post, rgrd, rblk, 0, stream,
                     hA, mx, hyp_b, pe_gamma, pe_beta, pos_embed, hidden);

  // 3. Mamba layers
  for (int i = 0; i < NDEPTH; i++) {
    hipLaunchKernelGGL(k_resid_ln, rgrd, rblk, 0, stream,
                       hidden, resid, gamma + (size_t)i * DM, beta + (size_t)i * DM,
                       hn, hn16, i == 0);
    gemm(stream, hn16, win16 + (size_t)i * 2 * DIN * DM, nullptr, xz, nullptr,
         MROWS, 2 * DIN, DM, DM, DM, 2 * DIN, ACT_NONE);
    hipLaunchKernelGGL(k_conv_silu, dim3((MROWS * DIN + 255) / 256), dim3(256), 0, stream,
                       xz, conv_w + (size_t)i * DIN * 4, conv_b + (size_t)i * DIN, u, u16);
    gemm(stream, u16, xp_pad16 + (size_t)i * 64 * DIN, nullptr, dbl, dbl16,
         MROWS, 64, DIN, DIN, DIN, DBLP, ACT_NONE);
    gemm(stream, dbl16, dt_pad16 + (size_t)i * DIN * 32, dt_proj_b + (size_t)i * DIN,
         dtb, nullptr, MROWS, DIN, 32, DBLP, 32, DIN, ACT_SOFTPLUS);
    hipLaunchKernelGGL(k_scan, dim3((BSZ * DIN + 255) / 256), dim3(256), 0, stream,
                       u, dtb, dbl, xz, A_log + (size_t)i * DIN * DST, Dp + (size_t)i * DIN, yb16);
    gemm(stream, yb16, wout16 + (size_t)i * DM * DIN, nullptr, hidden, nullptr,
         MROWS, DM, DIN, DIN, DIN, DM, ACT_NONE);
  }

  // 4. final norm, pool, classify
  hipLaunchKernelGGL(k_resid_ln, rgrd, rblk, 0, stream, hidden, resid, gamma_f, beta_f,
                     hn, hn16, 0);
  hipLaunchKernelGGL(k_logmap0_rows, rgrd, rblk, 0, stream, hn);
  hipLaunchKernelGGL(k_pool, dim3(BSZ), dim3(32), 0, stream, hn, pooled);
  hipLaunchKernelGGL(k_classifier, dim3((BSZ * NCLS + 255) / 256), dim3(256), 0, stream,
                     pooled, protos, out);
}
